// ArgumentGCN_5_9680856285904
// MI455X (gfx1250) — matrix operand order repaired and checked by reading, not compiled, not run on hardware
//
#include <hip/hip_runtime.h>
#include <hip/hip_bf16.h>
#include <math.h>

#define BB 32
#define NN 512
#define DD 256
#define STEPS 2

typedef __attribute__((ext_vector_type(16))) __bf16 v16bf;
typedef __attribute__((ext_vector_type(8)))  __bf16 v8bf;
typedef __attribute__((ext_vector_type(4)))  __bf16 v4bf;
typedef __attribute__((ext_vector_type(8)))  float  v8f;

struct GraphPtrs { const int* g[6]; };
struct WPtrs     { const float* w[7]; };

union AFrag { v16bf v; v8bf h[2]; };

// ---------------------------------------------------------------------------
// K0: transpose + convert the 7 weight matrices to bf16: WT[k][e][d] = W_k[d][e]
// ---------------------------------------------------------------------------
__global__ void k_convert_weights(WPtrs wp, __bf16* __restrict__ WT) {
  int idx = blockIdx.x * blockDim.x + threadIdx.x;
  if (idx >= 7 * DD * DD) return;
  int k = idx >> 16;            // DD*DD == 65536
  int rem = idx & 0xFFFF;
  int e = rem >> 8;
  int d = rem & 255;
  WT[idx] = (__bf16)wp.w[k][d * DD + e];
}

// ---------------------------------------------------------------------------
// K1: node f32 -> bf16 copy (A operand for the GEMMs)
// ---------------------------------------------------------------------------
__global__ void k_convert_node(const float* __restrict__ node, __bf16* __restrict__ nb) {
  int idx = blockIdx.x * blockDim.x + threadIdx.x;
  if (idx < BB * NN * DD) nb[idx] = (__bf16)node[idx];
}

// ---------------------------------------------------------------------------
// K2: inv_nn[b,m] = 1 / max(1, sum_k sum_n G_k*dd).  One wave per row.
// ---------------------------------------------------------------------------
__global__ void k_nn(GraphPtrs gp, const int* __restrict__ mask, float* __restrict__ inv_nn) {
  int row  = blockIdx.x * 8 + (threadIdx.x >> 5);   // b*NN + m
  int lane = threadIdx.x & 31;
  int b = row / NN, m = row % NN;
  const int* gr[6];
#pragma unroll
  for (int k = 0; k < 6; ++k) gr[k] = gp.g[k] + (size_t)row * NN;
  int cnt = 0;
  for (int n = lane; n < NN; n += 32) {
    if (n == m || !mask[b * NN + n]) continue;
    int c = 0;
#pragma unroll
    for (int k = 0; k < 6; ++k) c += (gr[k][n] != 0);
    cnt += c;
  }
#pragma unroll
  for (int s = 16; s > 0; s >>= 1) cnt += __shfl_xor(cnt, s, 32);
  if (lane == 0) {
    float nnf = (float)(mask[row] ? cnt : 0);
    float nnc = nnf >= 1.0f ? nnf : 1.0f;
    inv_nn[row] = 1.0f / nnc;
  }
}

// ---------------------------------------------------------------------------
// K3: d_node_weight = sigmoid(node . w_nw + b_nw).  One wave per row.
// ---------------------------------------------------------------------------
__global__ void k_dnw(const float* __restrict__ node, const float* __restrict__ w_nw,
                      const float* __restrict__ b_nw, const int* __restrict__ mask,
                      float* __restrict__ out_w, float* __restrict__ dnwm, int step) {
  int row  = blockIdx.x * 8 + (threadIdx.x >> 5);
  int lane = threadIdx.x & 31;
  const float4* nr = (const float4*)(node + (size_t)row * DD);
  const float4* wr = (const float4*)w_nw;
  float4 a0 = nr[lane * 2], a1 = nr[lane * 2 + 1];
  float4 b0 = wr[lane * 2], b1 = wr[lane * 2 + 1];
  float s = a0.x * b0.x + a0.y * b0.y + a0.z * b0.z + a0.w * b0.w
          + a1.x * b1.x + a1.y * b1.y + a1.z * b1.z + a1.w * b1.w;
#pragma unroll
  for (int d = 16; d > 0; d >>= 1) s += __shfl_xor(s, d, 32);
  if (lane == 0) {
    float sig = 1.0f / (1.0f + __expf(-(s + b_nw[0])));
    int b = row / NN, n = row % NN;
    out_w[(b * STEPS + step) * NN + n] = sig;
    dnwm[row] = mask[row] ? sig : 0.0f;
  }
}

// ---------------------------------------------------------------------------
// K4: projections.  blockIdx.x = b*7+k (k=0..5 -> projT bf16, k=6 -> selfinfo f32)
// ---------------------------------------------------------------------------
__global__ void __launch_bounds__(256)
k_proj(const __bf16* __restrict__ node_bf, const __bf16* __restrict__ WT,
       __bf16* __restrict__ projT, float* __restrict__ selfinfo,
       const float* __restrict__ b_self) {
  int bk = blockIdx.x;
  int b = bk / 7, k = bk % 7;
  int mtile = blockIdx.y;
  int w = threadIdx.x >> 5, lane = threadIdx.x & 31;
  int msub = w & 1, ebase = (w >> 1) * 64;
  int row0 = mtile * 32 + msub * 16;
  int lm = lane & 15, lh = lane >> 4;

  const __bf16* Abase = node_bf + (size_t)b * NN * DD + (size_t)(row0 + lm) * DD;
  const __bf16* WTk   = WT + (size_t)k * DD * DD;

  v8f zero = {0.f, 0.f, 0.f, 0.f, 0.f, 0.f, 0.f, 0.f};
  v8f acc[4];
#pragma unroll
  for (int et = 0; et < 4; ++et) acc[et] = zero;

#pragma unroll
  for (int d0 = 0; d0 < DD; d0 += 32) {
    AFrag a;
    int kh = lh * 8;
    a.h[0] = *(const v8bf*)(Abase + d0 + kh);
    a.h[1] = *(const v8bf*)(Abase + d0 + 16 + kh);
#pragma unroll
    for (int et = 0; et < 4; ++et) {
      int e = ebase + et * 16 + lm;
      const __bf16* bp = WTk + (size_t)e * DD + d0 + lh * 8;  // table-order K
      AFrag bb;
      bb.h[0] = *(const v8bf*)(bp);
      bb.h[1] = *(const v8bf*)(bp + 16);
      acc[et] = __builtin_amdgcn_wmma_f32_16x16x32_bf16(
          false, a.v, false, bb.v, (short)0, acc[et], false, false);
    }
  }

  if (k < 6) {
    __bf16* pb = projT + ((size_t)k * BB + b) * (size_t)DD * NN;
#pragma unroll
    for (int et = 0; et < 4; ++et) {
      int e = ebase + et * 16 + lm;
      v8bf o;
#pragma unroll
      for (int j = 0; j < 8; ++j) o[j] = (__bf16)acc[et][j];
      *(v8bf*)(pb + (size_t)e * NN + row0 + lh * 8) = o;
    }
  } else {
    float* si = selfinfo + (size_t)b * NN * DD;
#pragma unroll
    for (int et = 0; et < 4; ++et) {
      int e = ebase + et * 16 + lm;
      float bs = b_self[e];
#pragma unroll
      for (int j = 0; j < 8; ++j) {
        int n = row0 + lh * 8 + j;
        si[(size_t)n * DD + e] = acc[et][j] + bs;
      }
    }
  }
}

// ---------------------------------------------------------------------------
// K5: agg + fused update.  block = (b, 32-row tile), 256 thr = 8 waves.
//     Outer loop: 6 edge types (loop-invariant scalar graph/projT pointers).
//     Inner loop: 16 chunks of 32 neighbors, unroll-2 matching the ping-pong
//     LDS parity; ONE barrier per step; graph int4 prefetched one step ahead
//     (across the k6 boundary too) so HBM latency hides behind WMMAs.
// ---------------------------------------------------------------------------
__global__ void __launch_bounds__(256)
k_agg_update(GraphPtrs gp, const int* __restrict__ mask, const float* __restrict__ dnwm,
             const __bf16* __restrict__ projT, const float* __restrict__ selfinfo,
             const float* __restrict__ inv_nn, float* __restrict__ node_out,
             __bf16* __restrict__ node_bf_out) {
  int b = blockIdx.x;
  int mtile = blockIdx.y;
  int m0 = mtile * 32;
  int t = threadIdx.x;
  int w = t >> 5, lane = t & 31;
  int msub = w & 1, ebase = (w >> 1) * 64;
  int lm = lane & 15, lh = lane >> 4;

  __shared__ __align__(16) float  s_dnw[NN];
  __shared__ float s_rowm[32];
  __shared__ __align__(16) __bf16 s_A[2 * 32 * 40];  // ping-pong; rows padded to 40

  for (int n = t; n < NN; n += 256) s_dnw[n] = dnwm[b * NN + n];
  if (t < 32) s_rowm[t] = mask[b * NN + m0 + t] ? 1.0f : 0.0f;
  __syncthreads();

  v8f zero = {0.f, 0.f, 0.f, 0.f, 0.f, 0.f, 0.f, 0.f};
  v8f acc[4];
#pragma unroll
  for (int et = 0; et < 4; ++et) acc[et] = zero;

  int r  = t >> 3;         // builder row 0..31
  int c4 = (t & 7) * 4;    // builder col group
  int mg = m0 + r;
  int rloc = msub * 16 + lm;
  size_t row_off = (size_t)(b * NN + m0 + r) * NN + c4;
  float rm = s_rowm[r];    // hoisted: constant per thread

  const size_t pbStride = (size_t)BB * DD * NN;
  const __bf16* pbB = projT + (size_t)b * DD * NN;

  int4 g_next = *(const int4*)(gp.g[0] + row_off);

  for (int k6 = 0; k6 < 6; ++k6) {
    const int* grow = gp.g[k6] + row_off;
    const __bf16* pb = pbB + (size_t)k6 * pbStride;
#pragma unroll 2
    for (int nc = 0; nc < 16; ++nc) {
      int n0 = nc << 5;
      int4 g = g_next;
      // prefetch one step ahead (next chunk, or first chunk of next graph)
      if (nc < 15)      g_next = *(const int4*)(grow + n0 + 32);
      else if (k6 < 5)  g_next = *(const int4*)(gp.g[k6 + 1] + row_off);

      // ---- build masked A tile (32x32) into ping-pong buffer ----
      int nb0 = n0 + c4;
      v4bf av;
      av[0] = (__bf16)((g.x && (nb0     != mg)) ? s_dnw[nb0    ] * rm : 0.0f);
      av[1] = (__bf16)((g.y && (nb0 + 1 != mg)) ? s_dnw[nb0 + 1] * rm : 0.0f);
      av[2] = (__bf16)((g.z && (nb0 + 2 != mg)) ? s_dnw[nb0 + 2] * rm : 0.0f);
      av[3] = (__bf16)((g.w && (nb0 + 3 != mg)) ? s_dnw[nb0 + 3] * rm : 0.0f);
      __bf16* sA = s_A + (nc & 1) * (32 * 40);
      *(v4bf*)&sA[r * 40 + c4] = av;
      __syncthreads();   // tile visible; also guards buffer reuse two steps back

      // ---- fragments + WMMA ----
      AFrag a;
      a.h[0] = *(const v8bf*)&sA[rloc * 40 + lh * 8];
      a.h[1] = *(const v8bf*)&sA[rloc * 40 + 16 + lh * 8];
#pragma unroll
      for (int et = 0; et < 4; ++et) {
        int e = ebase + et * 16 + lm;
        const __bf16* bp = pb + (size_t)e * NN + n0 + lh * 8;
        AFrag bb;
        bb.h[0] = *(const v8bf*)(bp);
        bb.h[1] = *(const v8bf*)(bp + 16);
        acc[et] = __builtin_amdgcn_wmma_f32_16x16x32_bf16(
            false, a.v, false, bb.v, (short)0, acc[et], false, false);
      }
    }
  }

  // ---- fused update: relu(self + agg*inv_nn) ----
  float inv8[8];
#pragma unroll
  for (int j = 0; j < 8; ++j) inv8[j] = inv_nn[b * NN + m0 + msub * 16 + lh * 8 + j];
  const float* si = selfinfo    + (size_t)b * NN * DD;
  float*       no = node_out    + (size_t)b * NN * DD;
  __bf16*      nb = node_bf_out + (size_t)b * NN * DD;
#pragma unroll
  for (int et = 0; et < 4; ++et) {
    int e = ebase + et * 16 + lm;
#pragma unroll
    for (int j = 0; j < 8; ++j) {
      int mi = m0 + msub * 16 + lh * 8 + j;
      float v = si[(size_t)mi * DD + e] + acc[et][j] * inv8[j];
      v = v > 0.0f ? v : 0.0f;
      no[(size_t)mi * DD + e] = v;
      nb[(size_t)mi * DD + e] = (__bf16)v;
    }
  }
}

// ---------------------------------------------------------------------------
extern "C" void kernel_launch(void* const* d_in, const int* in_sizes, int n_in,
                              void* d_out, int out_size, void* d_ws, size_t ws_size,
                              hipStream_t stream) {
  (void)in_sizes; (void)n_in; (void)out_size; (void)ws_size;

  const float* node  = (const float*)d_in[0];
  const int*   mask  = (const int*)d_in[1];
  GraphPtrs gp;
  for (int i = 0; i < 6; ++i) gp.g[i] = (const int*)d_in[2 + i];
  const float* w_nw   = (const float*)d_in[8];
  const float* b_nw   = (const float*)d_in[9];
  const float* b_self = (const float*)d_in[11];
  WPtrs wp;
  for (int i = 0; i < 6; ++i) wp.w[i] = (const float*)d_in[12 + i];  // w_arg1..4, punct, punct_re
  wp.w[6] = (const float*)d_in[10];                                   // w_self last

  // workspace carve-up
  char* p = (char*)d_ws;
  auto carve = [&](size_t bytes) { char* r = p; p += (bytes + 255) & ~(size_t)255; return (void*)r; };
  __bf16* WT       = (__bf16*)carve((size_t)7 * DD * DD * 2);
  __bf16* node_bf  = (__bf16*)carve((size_t)BB * NN * DD * 2);
  __bf16* projT    = (__bf16*)carve((size_t)6 * BB * NN * DD * 2);
  float*  selfinfo = (float*) carve((size_t)BB * NN * DD * 4);
  float*  inv_nn   = (float*) carve((size_t)BB * NN * 4);
  float*  dnwm     = (float*) carve((size_t)BB * NN * 4);
  float*  node1    = (float*) carve((size_t)BB * NN * DD * 4);

  float* out_node = (float*)d_out;
  float* out_w    = out_node + (size_t)BB * NN * DD;

  k_convert_weights<<<(7 * DD * DD + 255) / 256, 256, 0, stream>>>(wp, WT);
  k_convert_node<<<(BB * NN * DD + 255) / 256, 256, 0, stream>>>(node, node_bf);
  k_nn<<<BB * NN / 8, 256, 0, stream>>>(gp, mask, inv_nn);

  const float* cur = node;
  for (int step = 0; step < STEPS; ++step) {
    k_dnw<<<BB * NN / 8, 256, 0, stream>>>(cur, w_nw, b_nw, mask, out_w, dnwm, step);
    k_proj<<<dim3(BB * 7, NN / 32), 256, 0, stream>>>(node_bf, WT, projT, selfinfo, b_self);
    float* nout = (step == STEPS - 1) ? out_node : node1;
    k_agg_update<<<dim3(BB, NN / 32), 256, 0, stream>>>(gp, mask, dnwm, projT, selfinfo,
                                                        inv_nn, nout, node_bf);
    cur = nout;
  }
}
